// Ge2eLoss_74835510165785
// MI455X (gfx1250) — matrix operand order repaired and checked by reading, not compiled, not run on hardware
//
#include <hip/hip_runtime.h>
#include <hip/hip_bf16.h>

// ---------------- problem constants ----------------
#define N_SPK   1024
#define M_UTT   32
#define D_EMB   512
#define NM      (N_SPK * M_UTT)          // 32768
#define INV_C   (1.0f / (1.0f - 32.0f))  // 1/(1-M) from reference
#define NPART   16                       // partials per row: 4 colblocks * 4 colgroups

typedef __attribute__((ext_vector_type(16))) __bf16 v16bf;
typedef __attribute__((ext_vector_type(8)))  __bf16 v8bf;
typedef __attribute__((ext_vector_type(4)))  __bf16 v4bf;
typedef __attribute__((ext_vector_type(2)))  __bf16 v2bf;
typedef __attribute__((ext_vector_type(8)))  float  v8f;

static __device__ __forceinline__ v16bf cat16(v8bf a, v8bf b) {
    return __builtin_shufflevector(a, b, 0,1,2,3,4,5,6,7,8,9,10,11,12,13,14,15);
}

// ---------------------------------------------------------------------------
// Kernel 1: per-speaker sums -> centroids (bf16 hi/lo) + exact fp32 diagonal
// grid = N_SPK blocks, 256 threads
// ---------------------------------------------------------------------------
__global__ __launch_bounds__(256) void ge2e_reduce_kernel(
    const float* __restrict__ e, __bf16* __restrict__ cenHi,
    __bf16* __restrict__ cenLo, float* __restrict__ diag)
{
    __shared__ float sums[D_EMB];
    const int n = blockIdx.x;
    const int t = threadIdx.x;
    const float* base = e + (size_t)n * M_UTT * D_EMB;

    // sum over utterances; thread t owns columns 2t, 2t+1
    float sx = 0.f, sy = 0.f;
    #pragma unroll 8
    for (int m = 0; m < M_UTT; ++m) {
        const float2 v = *(const float2*)(base + m * D_EMB + t * 2);
        sx += v.x; sy += v.y;
    }
    sums[2 * t]     = sx;
    sums[2 * t + 1] = sy;

    // centroid = sum / M, stored as split bf16 (hi + lo ~ fp32)
    const float cx = sx * (1.0f / 32.0f);
    const float cy = sy * (1.0f / 32.0f);
    const __bf16 hx = (__bf16)cx, hy = (__bf16)cy;
    const __bf16 lx = (__bf16)(cx - (float)hx), ly = (__bf16)(cy - (float)hy);
    v2bf hv = {hx, hy}, lv = {lx, ly};
    *(v2bf*)(cenHi + (size_t)n * D_EMB + 2 * t) = hv;
    *(v2bf*)(cenLo + (size_t)n * D_EMB + 2 * t) = lv;
    __syncthreads();

    // diag[n,m] = e.sum - inv * e.e   (exact fp32)
    const int wave = t >> 5, lane = t & 31;
    #pragma unroll
    for (int i = 0; i < 4; ++i) {
        const int m = wave * 4 + i;
        const float* row = base + m * D_EMB;
        float s1 = 0.f, s2 = 0.f;
        #pragma unroll 4
        for (int j = 0; j < 16; ++j) {
            const int d = lane + j * 32;
            const float x = row[d];
            s1 += x * sums[d];
            s2 += x * x;
        }
        #pragma unroll
        for (int mask = 16; mask >= 1; mask >>= 1) {
            s1 += __shfl_xor(s1, mask);
            s2 += __shfl_xor(s2, mask);
        }
        if (lane == 0) diag[n * M_UTT + m] = s1 - INV_C * s2;
    }
}

// ---------------------------------------------------------------------------
// Kernel 2: split-bf16 WMMA GEMM (rows x centroids^T) fused with diagonal
// overwrite and partial logsumexp.
// grid = (NM/32, 4); block = 256 (8 waves = 2 row-slices x 4 col-groups)
// Each WG: rows [rb*32, rb*32+32) (exactly speaker rb) x cols [cb*256,+256)
// Three independent accumulator chains (HH, HL, LH) keep the XDL pipe full.
// ---------------------------------------------------------------------------
__global__ __launch_bounds__(256) void ge2e_gemm_lse_kernel(
    const float* __restrict__ e, const __bf16* __restrict__ cenHi,
    const __bf16* __restrict__ cenLo, const float* __restrict__ diag,
    const float* __restrict__ wp, const float* __restrict__ bp,
    float2* __restrict__ part)
{
    // A fragments, swizzled to the ISA 16-bit A-matrix layout:
    // idx = (((slice*16 + kc)*32 + lane)*2 + run)*8 + pos
    __shared__ __bf16 ah[32 * D_EMB];   // 32 KB
    __shared__ __bf16 al[32 * D_EMB];   // 32 KB

    const int rb = blockIdx.x;          // 32-row block == speaker id == label col
    const int cb = blockIdx.y;          // 256-col block
    const int t  = threadIdx.x;

    // ---- stage A: fp32 -> bf16 hi/lo, fragment-swizzled in LDS ----
    const float4* src = (const float4*)(e + (size_t)rb * 32 * D_EMB);
    #pragma unroll
    for (int i = 0; i < 16; ++i) {
        const int id  = i * 256 + t;
        const int row = id >> 7;           // 0..31
        const int k4  = id & 127;          // float4 index within row
        const float4 v = src[row * 128 + k4];
        const int k     = k4 * 4;
        const int slice = row >> 4, r = row & 15;
        const int kc    = k >> 5,   kk = k & 31;
        const int lane  = r + (((kk >> 3) & 1) << 4);
        const int run   = kk >> 4,  pos = kk & 7;
        const int bidx  = (((slice * 16 + kc) * 32 + lane) * 2 + run) * 8 + pos;
        const __bf16 h0 = (__bf16)v.x, h1 = (__bf16)v.y,
                     h2 = (__bf16)v.z, h3 = (__bf16)v.w;
        v4bf hv = {h0, h1, h2, h3};
        v4bf lv = {(__bf16)(v.x - (float)h0), (__bf16)(v.y - (float)h1),
                   (__bf16)(v.z - (float)h2), (__bf16)(v.w - (float)h3)};
        *(v4bf*)&ah[bidx] = hv;
        *(v4bf*)&al[bidx] = lv;
    }
    __syncthreads();

    const float w = *wp, b = *bp;
    const int wave = t >> 5, lane = t & 31;
    const int slice = wave & 1;         // which 16-row slice
    const int cg    = wave >> 1;        // which col-group (4 tiles of 16 cols)
    const int rowbase = rb * 32 + slice * 16;
    const int label   = rb;             // label column for every row in this block
    const int halfsel = lane >> 4;      // 0: lanes 0-15, 1: lanes 16-31

    float runM[8], runS[8];
    #pragma unroll
    for (int v = 0; v < 8; ++v) { runM[v] = -3.0e38f; runS[v] = 0.f; }

    for (int it = 0; it < 4; ++it) {
        const int colbase = cb * 256 + (cg + it * 4) * 16;
        // B fragment source: column (colbase + lane&15) == centroid row;
        // lanes 0-15 take K 0..7,16..23 of each chunk, lanes 16-31 take K 8..15,24..31.
        const size_t boff = (size_t)(colbase + (lane & 15)) * D_EMB + halfsel * 8;
        const __bf16* bhp = cenHi + boff;
        const __bf16* blp = cenLo + boff;

        v8f accHH = {};   // Ahi x Bhi
        v8f accHL = {};   // Ahi x Blo
        v8f accLH = {};   // Alo x Bhi
        #pragma unroll 4
        for (int kc = 0; kc < 16; ++kc) {
            const int abase = ((slice * 16 + kc) * 32 + lane) * 16;
            const v16bf aHi = cat16(*(const v8bf*)&ah[abase],
                                    *(const v8bf*)&ah[abase + 8]);
            const v16bf aLo = cat16(*(const v8bf*)&al[abase],
                                    *(const v8bf*)&al[abase + 8]);
            const __bf16* b0 = bhp + kc * 32;
            const __bf16* b1 = blp + kc * 32;
            const v16bf bHi = cat16(*(const v8bf*)b0, *(const v8bf*)(b0 + 16));
            const v16bf bLo = cat16(*(const v8bf*)b1, *(const v8bf*)(b1 + 16));
            // (Ahi+Alo)x(Bhi+Blo) minus lo*lo term; 3 independent chains
            accHH = __builtin_amdgcn_wmma_f32_16x16x32_bf16(false, aHi, false, bHi,
                                                            (short)0, accHH, false, false);
            accHL = __builtin_amdgcn_wmma_f32_16x16x32_bf16(false, aHi, false, bLo,
                                                            (short)0, accHL, false, false);
            accLH = __builtin_amdgcn_wmma_f32_16x16x32_bf16(false, aLo, false, bHi,
                                                            (short)0, accLH, false, false);
        }

        // epilogue: combine chains, logits, diagonal overwrite, tile logsumexp
        const bool myCol = (label >= colbase) && (label < colbase + 16) &&
                           ((lane & 15) == (label & 15));
        const int mrow = rowbase + 8 * halfsel;   // + v gives global row
        float vals[8];
        #pragma unroll
        for (int v = 0; v < 8; ++v) {
            const float dot = accHH[v] + (accHL[v] + accLH[v]);
            float x = w * dot + b;
            if (myCol) x = w * diag[mrow + v] + b;   // exact label logit
            vals[v] = x;
        }
        #pragma unroll
        for (int v = 0; v < 8; ++v) {
            float m = vals[v];
            #pragma unroll
            for (int mask = 8; mask >= 1; mask >>= 1)
                m = fmaxf(m, __shfl_xor(m, mask));       // 16-lane half reduce
            float s = __expf(vals[v] - m);
            #pragma unroll
            for (int mask = 8; mask >= 1; mask >>= 1)
                s += __shfl_xor(s, mask);
            const float nm = fmaxf(runM[v], m);
            runS[v] = runS[v] * __expf(runM[v] - nm) + s * __expf(m - nm);
            runM[v] = nm;
        }
    }

    // lanes 0 and 16 hold the reduced (max,sum) for their 8 rows
    if ((lane & 15) == 0) {
        const int rbase2 = rowbase + 8 * halfsel;
        const int pidx   = cb * 4 + cg;
        #pragma unroll
        for (int v = 0; v < 8; ++v) {
            float2 q; q.x = runM[v]; q.y = runS[v];
            part[(size_t)(rbase2 + v) * NPART + pidx] = q;
        }
    }
}

// ---------------------------------------------------------------------------
// Kernel 3: combine partials -> logz, loss = mean(logz - label_logit)
// grid = NM/256 blocks
// ---------------------------------------------------------------------------
__global__ __launch_bounds__(256) void ge2e_final_kernel(
    const float2* __restrict__ part, const float* __restrict__ diag,
    const float* __restrict__ wp, const float* __restrict__ bp,
    float* __restrict__ out)
{
    __shared__ float red[256];
    const int r = blockIdx.x * 256 + threadIdx.x;
    const float w = *wp, b = *bp;

    float M = -3.0e38f, S = 0.f;
    #pragma unroll
    for (int p = 0; p < NPART; ++p) {
        const float2 q = part[(size_t)r * NPART + p];
        const float nm = fmaxf(M, q.x);
        S = S * __expf(M - nm) + q.y * __expf(q.x - nm);
        M = nm;
    }
    const float logz = M + __logf(S);
    const float contrib = logz - (w * diag[r] + b);

    red[threadIdx.x] = contrib;
    __syncthreads();
    #pragma unroll
    for (int s = 128; s > 0; s >>= 1) {
        if (threadIdx.x < s) red[threadIdx.x] += red[threadIdx.x + s];
        __syncthreads();
    }
    if (threadIdx.x == 0) atomicAdd(out, red[0] * (1.0f / (float)NM));
}

// ---------------------------------------------------------------------------
extern "C" void kernel_launch(void* const* d_in, const int* in_sizes, int n_in,
                              void* d_out, int out_size, void* d_ws, size_t ws_size,
                              hipStream_t stream)
{
    const float* e  = (const float*)d_in[0];
    const float* wv = (const float*)d_in[1];
    const float* bv = (const float*)d_in[2];
    float* out = (float*)d_out;

    // workspace layout (~6.4 MB)
    char* ws = (char*)d_ws;
    __bf16* cenHi = (__bf16*)ws;                              // 1 MB
    __bf16* cenLo = (__bf16*)(ws + (size_t)(1u << 20));       // 1 MB
    float*  diag  = (float*)(ws + (size_t)(2u << 20));        // 128 KB
    float2* part  = (float2*)(ws + (size_t)(2u << 20) + NM * sizeof(float)); // 4 MB

    hipMemsetAsync(d_out, 0, sizeof(float), stream);

    ge2e_reduce_kernel<<<N_SPK, 256, 0, stream>>>(e, cenHi, cenLo, diag);

    dim3 g2(NM / 32, 4);
    ge2e_gemm_lse_kernel<<<g2, 256, 0, stream>>>(e, cenHi, cenLo, diag, wv, bv, part);

    ge2e_final_kernel<<<NM / 256, 256, 0, stream>>>(part, diag, wv, bv, out);
}
